// SparseSelfAttention_79156247265919
// MI455X (gfx1250) — matrix operand order repaired and checked by reading, not compiled, not run on hardware
//
#include <hip/hip_runtime.h>
#include <hip/hip_bf16.h>
#include <math.h>

// Problem constants (match reference)
#define BB 4
#define SS 2048
#define DD 1024
#define EE 8
#define HDM 128           // head dim per expert
#define NEB (EE*BB)       // 32 (expert,batch) pairs

typedef __bf16 bf16_t;
typedef __bf16 bf16x16 __attribute__((ext_vector_type(16)));
typedef float  f32x8   __attribute__((ext_vector_type(8)));

union Frag16 {            // 16 bf16 elements = 8 dwords
    bf16x16        v;
    unsigned int   u[8];
    unsigned short h[16];
};

__device__ __forceinline__ unsigned short bfbits(float f) {
    union { bf16_t b; unsigned short u; } cv; cv.b = (bf16_t)f; return cv.u;
}

__device__ __forceinline__ unsigned int packbf2(float lo, float hi) {
    return (unsigned int)bfbits(lo) | ((unsigned int)bfbits(hi) << 16);
}

// A-matrix (16x32 bf16) K index for vgpr v, lane-half h (contiguous: k = 16h + 2v, matches B)
__device__ __forceinline__ int amap_k(int v, int half) {
    return half * 16 + 2 * v;
}

__device__ __forceinline__ f32x8 wmma_bf(bf16x16 a, bf16x16 b, f32x8 c) {
    return __builtin_amdgcn_wmma_f32_16x16x32_bf16(false, a, false, b, (short)0, c,
                                                   false, false);
}

// ---------------------------------------------------------------------------
// K0: fp32 -> bf16 weight conversion, TRANSPOSED so WMMA B-fragments are
// k-contiguous (2 x b128 per fragment instead of 16 x u16):
//   WqkvT[e][n(384)][k(1024)],  WffT[e][n(1024)][k(128)]
// ---------------------------------------------------------------------------
__global__ void k_cvt_weights(const float* __restrict__ Wqkv,
                              const float* __restrict__ Wff,
                              bf16_t* __restrict__ WqkvT,
                              bf16_t* __restrict__ WffT) {
    const int n1 = EE * 384 * DD;       // 3,145,728
    const int n2 = EE * DD * HDM;       // 1,048,576
    int i = blockIdx.x * blockDim.x + threadIdx.x;
    if (i < n1) {
        int e = i / (384 * DD), r = i % (384 * DD);
        int n = r / DD, k = r % DD;
        WqkvT[i] = (bf16_t)Wqkv[(size_t)e * DD * 384 + (size_t)k * 384 + n];
    } else if (i - n1 < n2) {
        int j = i - n1;
        int e = j / (DD * HDM), r = j % (DD * HDM);
        int n = r / HDM, k = r % HDM;
        WffT[j] = (bf16_t)Wff[(size_t)e * HDM * DD + (size_t)k * DD + n];
    }
}

// ---------------------------------------------------------------------------
// K1: fused RoPE + gate. One wave per token (wave32). Lane covers 32 dims.
// ---------------------------------------------------------------------------
__global__ void k_rope_gate(const float* __restrict__ x,
                            const float* __restrict__ Wg,
                            const float* __restrict__ cosb,
                            const float* __restrict__ sinb,
                            bf16_t* __restrict__ xh_bf,
                            float* __restrict__ gate,
                            int* __restrict__ expert) {
    const int wave = threadIdx.x >> 5;
    const int lane = threadIdx.x & 31;
    const int t = blockIdx.x * 8 + wave;        // token 0..B*S-1
    const int s = t & (SS - 1);
    const float* xp = x + (size_t)t * DD;

    float acc[EE];
#pragma unroll
    for (int e = 0; e < EE; ++e) acc[e] = 0.f;

    for (int j = 0; j < 32; ++j) {
        int d = j * 32 + lane;
        int h = d >> 7, r = d & 127;
        float val;
        if (r < 64) {                           // rotated PE half goes first
            int i = r >> 1;
            float re = xp[h * 128 + 64 + 2 * i];
            float im = xp[h * 128 + 64 + 2 * i + 1];
            float c  = cosb[s * 32 + i];
            float sn = sinb[s * 32 + i];
            val = ((r & 1) == 0) ? (re * c - im * sn) : (re * sn + im * c);
        } else {
            val = xp[h * 128 + (r - 64)];       // nope half moved up
        }
        xh_bf[(size_t)t * DD + d] = (bf16_t)val;
        const float* wr = Wg + (size_t)d * EE;
#pragma unroll
        for (int e = 0; e < EE; ++e) acc[e] += val * wr[e];
    }
#pragma unroll
    for (int e = 0; e < EE; ++e) {
        float a = acc[e];
        a += __shfl_xor(a, 16, 32);
        a += __shfl_xor(a, 8, 32);
        a += __shfl_xor(a, 4, 32);
        a += __shfl_xor(a, 2, 32);
        a += __shfl_xor(a, 1, 32);
        acc[e] = a;
    }
    float lmax = acc[0]; int arg = 0;
#pragma unroll
    for (int e = 1; e < EE; ++e) if (acc[e] > lmax) { lmax = acc[e]; arg = e; }
    float sum = 0.f;
#pragma unroll
    for (int e = 0; e < EE; ++e) sum += __expf(acc[e] - lmax);
    if (lane == 0) { gate[t] = 1.0f / sum; expert[t] = arg; }
}

// ---------------------------------------------------------------------------
// K2: stable compaction per (e,b): idxlist[eb][0..Ne) = tokens of expert e
// ---------------------------------------------------------------------------
__global__ void k_route(const int* __restrict__ expert,
                        int* __restrict__ idxlist,
                        int* __restrict__ NeArr) {
    const int eb = blockIdx.x;
    const int e = eb / BB, b = eb % BB;
    __shared__ int sc[256];
    const int tid = threadIdx.x;
    int f[8], cnt = 0;
    const int base = tid * 8;
#pragma unroll
    for (int j = 0; j < 8; ++j) {
        f[j] = (expert[b * SS + base + j] == e);
        cnt += f[j];
    }
    sc[tid] = cnt;
    __syncthreads();
    for (int off = 1; off < 256; off <<= 1) {   // inclusive Hillis-Steele scan
        int v = (tid >= off) ? sc[tid - off] : 0;
        __syncthreads();
        sc[tid] += v;
        __syncthreads();
    }
    int pos = sc[tid] - cnt;                    // exclusive prefix
#pragma unroll
    for (int j = 0; j < 8; ++j)
        if (f[j]) idxlist[(size_t)eb * SS + pos++] = base + j;
    if (tid == 255) NeArr[eb] = sc[255];
}

// ---------------------------------------------------------------------------
// K3: QKV GEMM per (e,b):  (Ne x 1024) @ (1024 x 384) -> Q,K row-major bf16;
// V stored TRANSPOSED (Vt[eb][feat][token]) so the P*V B-fragment streams
// contiguously. grid (EB, S/16), block 256 (8 waves), wave = 3 of 24 N-tiles.
// ---------------------------------------------------------------------------
__global__ void k_qkv(const bf16_t* __restrict__ xh_bf,
                      const bf16_t* __restrict__ WqkvT,
                      const int* __restrict__ idxlist,
                      const int* __restrict__ NeArr,
                      bf16_t* __restrict__ Qbf,
                      bf16_t* __restrict__ Kbf,
                      bf16_t* __restrict__ Vt) {
    const int eb = blockIdx.x;
    const int Ne = NeArr[eb];
    const int mtile = blockIdx.y;
    if (mtile * 16 >= Ne) return;
    const int e = eb / BB, b = eb % BB;
    const int wave = threadIdx.x >> 5, lane = threadIdx.x & 31;
    const int half = lane >> 4, l16 = lane & 15;

    int arow = mtile * 16 + l16;
    if (arow >= Ne) arow = Ne - 1;              // clamp; garbage rows not stored
    const int tokA = idxlist[(size_t)eb * SS + arow];
    const bf16_t* aBase = xh_bf + ((size_t)b * SS + tokA) * DD;
    // B base per N-column: WqkvT + (e*384 + col) * 1024, k contiguous
    const bf16_t* wq = WqkvT + (size_t)e * 384 * DD;

    const f32x8 zero8 = {0, 0, 0, 0, 0, 0, 0, 0};
    f32x8 acc[3] = {zero8, zero8, zero8};
    const int col0 = (wave * 3) * 16 + l16;
    const bf16_t* bBase0 = wq + (size_t)(col0      ) * DD;
    const bf16_t* bBase1 = wq + (size_t)(col0 +  16) * DD;
    const bf16_t* bBase2 = wq + (size_t)(col0 +  32) * DD;

    for (int kc = 0; kc < 32; ++kc) {
        __builtin_prefetch(aBase + (kc + 2) * 32, 0, 0);   // stream-ahead on A
        Frag16 a, b0, b1, b2;
        const int ka = kc * 32;
        const int kb = kc * 32 + half * 16;
#pragma unroll
        for (int v = 0; v < 8; ++v) {
            a.u[v]  = *(const unsigned int*)(aBase  + ka + amap_k(v, half));
            b0.u[v] = *(const unsigned int*)(bBase0 + kb + 2 * v);
            b1.u[v] = *(const unsigned int*)(bBase1 + kb + 2 * v);
            b2.u[v] = *(const unsigned int*)(bBase2 + kb + 2 * v);
        }
        acc[0] = wmma_bf(a.v, b0.v, acc[0]);
        acc[1] = wmma_bf(a.v, b1.v, acc[1]);
        acc[2] = wmma_bf(a.v, b2.v, acc[2]);
    }
#pragma unroll
    for (int t = 0; t < 3; ++t) {
        const int col = (wave * 3 + t) * 16 + l16;
        if (col < 256) {                         // Q / K: row-major stores
            bf16_t* dst = (col < 128) ? Qbf : Kbf;
            const int feat = col & 127;
#pragma unroll
            for (int v = 0; v < 8; ++v) {
                int row = mtile * 16 + v + 8 * half;
                if (row < Ne)
                    dst[((size_t)eb * SS + row) * HDM + feat] = (bf16_t)acc[t][v];
            }
        } else {                                 // V: transposed, rows contiguous
            const int feat = col - 256;
            bf16_t* dst = Vt + ((size_t)eb * HDM + feat) * SS;
#pragma unroll
            for (int v = 0; v < 8; v += 2) {
                int row = mtile * 16 + v + 8 * half;          // even
                if (row + 1 < Ne) {
                    *(unsigned int*)(dst + row) = packbf2(acc[t][v], acc[t][v + 1]);
                } else if (row < Ne) {
                    dst[row] = (bf16_t)acc[t][v];
                }
            }
        }
    }
}

// ---------------------------------------------------------------------------
// K4: flash attention per (e,b, 16-query tile). One wave per block.
// Row stats live per-VGPR (row -> (v, lane-half) of the C/D fragment).
// P staged through 2KB LDS to re-fragment as WMMA-A.
// ---------------------------------------------------------------------------
__global__ void k_attn(const bf16_t* __restrict__ Qbf,
                       const bf16_t* __restrict__ Kbf,
                       const bf16_t* __restrict__ Vt,
                       const int* __restrict__ NeArr,
                       bf16_t* __restrict__ Ctxbf) {
    const int eb = blockIdx.x;
    const int Ne = NeArr[eb];
    const int qtile = blockIdx.y;
    if (qtile * 16 >= Ne) return;
    const int lane = threadIdx.x & 31;
    const int half = lane >> 4, l16 = lane & 15;
    __shared__ float psh[16 * 33];

    const bf16_t* qb = Qbf + ((size_t)eb * SS + qtile * 16 + l16) * HDM;
    Frag16 qf[4];
#pragma unroll
    for (int kc = 0; kc < 4; ++kc)
#pragma unroll
        for (int v = 0; v < 8; ++v)
            qf[kc].u[v] = *(const unsigned int*)(qb + kc * 32 + amap_k(v, half));

    const f32x8 zero8 = {0, 0, 0, 0, 0, 0, 0, 0};
    f32x8 ctx[8];
    float m_run[8], l_run[8];
#pragma unroll
    for (int t = 0; t < 8; ++t) ctx[t] = zero8;
#pragma unroll
    for (int v = 0; v < 8; ++v) { m_run[v] = -1e30f; l_run[v] = 0.f; }

    const float scale = 0.0883883476483184f;    // 1/sqrt(128)
    const int nkt = (Ne + 31) >> 5;             // key tiles of 32
    const bf16_t* vtb = Vt + (size_t)eb * HDM * SS;

    for (int j = 0; j < nkt; ++j) {
        const int kb = j * 32;
        f32x8 s0 = zero8, s1 = zero8;
        const bf16_t* k0 = Kbf + ((size_t)eb * SS + kb + l16) * HDM;
        const bf16_t* k1 = k0 + 16 * HDM;
#pragma unroll
        for (int kc = 0; kc < 4; ++kc) {
            Frag16 b0, b1;
#pragma unroll
            for (int v = 0; v < 8; ++v) {
                int kk = kc * 32 + half * 16 + 2 * v;
                b0.u[v] = *(const unsigned int*)(k0 + kk);
                b1.u[v] = *(const unsigned int*)(k1 + kk);
            }
            s0 = wmma_bf(qf[kc].v, b0.v, s0);
            s1 = wmma_bf(qf[kc].v, b1.v, s1);
        }
        const bool val0 = (kb + l16) < Ne;
        const bool val1 = (kb + 16 + l16) < Ne;
        float corr[8];
#pragma unroll
        for (int v = 0; v < 8; ++v) {
            s0[v] = val0 ? s0[v] * scale : -1e30f;
            s1[v] = val1 ? s1[v] * scale : -1e30f;
            float m = fmaxf(s0[v], s1[v]);      // row max within 16-lane group
            m = fmaxf(m, __shfl_xor(m, 8, 32));
            m = fmaxf(m, __shfl_xor(m, 4, 32));
            m = fmaxf(m, __shfl_xor(m, 2, 32));
            m = fmaxf(m, __shfl_xor(m, 1, 32));
            float nm = fmaxf(m_run[v], m);
            corr[v] = __expf(m_run[v] - nm);
            m_run[v] = nm;
            float p0 = __expf(s0[v] - nm);
            float p1 = __expf(s1[v] - nm);
            s0[v] = p0; s1[v] = p1;
            float rs = p0 + p1;
            rs += __shfl_xor(rs, 8, 32);
            rs += __shfl_xor(rs, 4, 32);
            rs += __shfl_xor(rs, 2, 32);
            rs += __shfl_xor(rs, 1, 32);
            l_run[v] = l_run[v] * corr[v] + rs;
        }
#pragma unroll
        for (int t = 0; t < 8; ++t)
#pragma unroll
            for (int v = 0; v < 8; ++v) ctx[t][v] *= corr[v];

        __syncthreads();
#pragma unroll
        for (int v = 0; v < 8; ++v) {
            int row = v + 8 * half;             // D-fragment row
            psh[row * 33 + l16]      = s0[v];
            psh[row * 33 + 16 + l16] = s1[v];
        }
        __syncthreads();
        Frag16 pa;
#pragma unroll
        for (int v = 0; v < 8; ++v) {
            int k = amap_k(v, half);
            pa.h[2 * v]     = bfbits(psh[l16 * 33 + k]);
            pa.h[2 * v + 1] = bfbits(psh[l16 * 33 + k + 1]);
        }
#pragma unroll
        for (int t = 0; t < 8; ++t) {
            // Vt: feat-major, keys contiguous -> 8 contiguous dwords
            const bf16_t* vrow = vtb + (size_t)(t * 16 + l16) * SS + kb + half * 16;
            Frag16 bv;
#pragma unroll
            for (int v = 0; v < 8; ++v)
                bv.u[v] = *(const unsigned int*)(vrow + 2 * v);
            ctx[t] = wmma_bf(pa.v, bv.v, ctx[t]);
        }
    }
#pragma unroll
    for (int t = 0; t < 8; ++t)
#pragma unroll
        for (int v = 0; v < 8; ++v) {
            int row = qtile * 16 + v + 8 * half;
            if (row < Ne)
                Ctxbf[((size_t)eb * SS + row) * HDM + t * 16 + l16] =
                    (bf16_t)(ctx[t][v] / l_run[v]);
        }
}

// ---------------------------------------------------------------------------
// K5: out = (ctx @ Wff[e] + b_ff) * gate, exclusive scatter to d_out
// grid (EB, S/16), block 256 (8 waves), wave handles 8 of 64 N-tiles
// ---------------------------------------------------------------------------
__global__ void k_out(const bf16_t* __restrict__ Ctxbf,
                      const bf16_t* __restrict__ WffT,
                      const float* __restrict__ b_ff,
                      const float* __restrict__ gate,
                      const int* __restrict__ idxlist,
                      const int* __restrict__ NeArr,
                      float* __restrict__ out) {
    const int eb = blockIdx.x;
    const int Ne = NeArr[eb];
    const int mtile = blockIdx.y;
    if (mtile * 16 >= Ne) return;
    const int e = eb / BB, b = eb % BB;
    const int wave = threadIdx.x >> 5, lane = threadIdx.x & 31;
    const int half = lane >> 4, l16 = lane & 15;

    const bf16_t* aBase = Ctxbf + ((size_t)eb * SS + mtile * 16 + l16) * HDM;
    const bf16_t* wf = WffT + (size_t)e * DD * HDM;   // [n(1024)][k(128)]

    const f32x8 zero8 = {0, 0, 0, 0, 0, 0, 0, 0};
    f32x8 acc[8];
#pragma unroll
    for (int t = 0; t < 8; ++t) acc[t] = zero8;

#pragma unroll
    for (int kc = 0; kc < 4; ++kc) {
        __builtin_prefetch(aBase + (kc + 1) * 32, 0, 0);
        Frag16 a;
#pragma unroll
        for (int v = 0; v < 8; ++v)
            a.u[v] = *(const unsigned int*)(aBase + kc * 32 + amap_k(v, half));
#pragma unroll
        for (int t = 0; t < 8; ++t) {
            const int col = (wave * 8 + t) * 16 + l16;
            const bf16_t* bBase = wf + (size_t)col * HDM + kc * 32 + half * 16;
            Frag16 bw;
#pragma unroll
            for (int v = 0; v < 8; ++v)
                bw.u[v] = *(const unsigned int*)(bBase + 2 * v);
            acc[t] = wmma_bf(a.v, bw.v, acc[t]);
        }
    }
    int tok[8]; float g[8];
#pragma unroll
    for (int v = 0; v < 8; ++v) {
        int row = mtile * 16 + v + 8 * half;
        int r = (row < Ne) ? row : 0;
        tok[v] = idxlist[(size_t)eb * SS + r];
        g[v]   = gate[b * SS + tok[v]];
    }
#pragma unroll
    for (int t = 0; t < 8; ++t) {
        const int col = (wave * 8 + t) * 16 + l16;
        const float bias = b_ff[col];
#pragma unroll
        for (int v = 0; v < 8; ++v) {
            int row = mtile * 16 + v + 8 * half;
            if (row < Ne)
                out[((size_t)b * SS + tok[v]) * DD + col] = (acc[t][v] + bias) * g[v];
        }
    }
}

// ---------------------------------------------------------------------------
extern "C" void kernel_launch(void* const* d_in, const int* in_sizes, int n_in,
                              void* d_out, int out_size, void* d_ws, size_t ws_size,
                              hipStream_t stream) {
    const float* x    = (const float*)d_in[0];
    const float* Wg   = (const float*)d_in[1];
    const float* Wqkv = (const float*)d_in[2];
    const float* Wff  = (const float*)d_in[3];
    const float* b_ff = (const float*)d_in[4];
    const float* cosb = (const float*)d_in[5];
    const float* sinb = (const float*)d_in[6];
    float* out = (float*)d_out;

    char* ws = (char*)d_ws;
    size_t off = 0;
    auto carve = [&](size_t bytes) {
        void* p = ws + off;
        off += (bytes + 255) & ~(size_t)255;
        return p;
    };
    bf16_t* WqkvT  = (bf16_t*)carve((size_t)EE * 384 * DD * 2);
    bf16_t* WffT   = (bf16_t*)carve((size_t)EE * DD * HDM * 2);
    bf16_t* xh_bf  = (bf16_t*)carve((size_t)BB * SS * DD * 2);
    float*  gate   = (float*) carve((size_t)BB * SS * 4);
    int*    expert = (int*)   carve((size_t)BB * SS * 4);
    int*    idxlist= (int*)   carve((size_t)NEB * SS * 4);
    int*    NeArr  = (int*)   carve((size_t)NEB * 4);
    bf16_t* Qbf    = (bf16_t*)carve((size_t)NEB * SS * HDM * 2);
    bf16_t* Kbf    = (bf16_t*)carve((size_t)NEB * SS * HDM * 2);
    bf16_t* Vt     = (bf16_t*)carve((size_t)NEB * HDM * SS * 2);
    bf16_t* Ctxbf  = (bf16_t*)carve((size_t)NEB * SS * HDM * 2);
    (void)in_sizes; (void)n_in; (void)out_size; (void)ws_size;

    const int ncvt = EE * 384 * DD + EE * DD * HDM;
    k_cvt_weights<<<(ncvt + 255) / 256, 256, 0, stream>>>(Wqkv, Wff, WqkvT, WffT);
    k_rope_gate<<<BB * SS / 8, 256, 0, stream>>>(x, Wg, cosb, sinb, xh_bf, gate, expert);
    k_route<<<NEB, 256, 0, stream>>>(expert, idxlist, NeArr);
    k_qkv<<<dim3(NEB, SS / 16), 256, 0, stream>>>(xh_bf, WqkvT, idxlist, NeArr,
                                                  Qbf, Kbf, Vt);
    k_attn<<<dim3(NEB, SS / 16), 32, 0, stream>>>(Qbf, Kbf, Vt, NeArr, Ctxbf);
    k_out<<<dim3(NEB, SS / 16), 256, 0, stream>>>(Ctxbf, WffT, b_ff, gate, idxlist,
                                                  NeArr, out);
}
